// TransformerAttLayerRankQ_2576980378154
// MI455X (gfx1250) — compile-verified, matrix operand order checked by reading, not run on hardware
//
#include <hip/hip_runtime.h>
#include <stdint.h>

// ---------- types ----------
typedef __attribute__((ext_vector_type(16))) __bf16   bf16x16;
typedef __attribute__((ext_vector_type(8)))  float    f32x8;
typedef __attribute__((ext_vector_type(4)))  unsigned u32x4;

union Frag { u32x4 q[2]; bf16x16 v; };

__device__ inline float b2f(uint16_t h) {
    union { unsigned u; float f; } c; c.u = ((unsigned)h) << 16; return c.f;
}
__device__ inline uint16_t f2b(float f) {
    union { float f; unsigned u; } c; c.f = f;
    unsigned u = c.u + 0x7FFFu + ((c.u >> 16) & 1u);   // round-to-nearest-even
    return (uint16_t)(u >> 16);
}

struct MabW {
    const uint16_t *Wq, *Wk, *Wv, *Wo;   // transposed bf16 [N=128][Kpad]
    const float *bq, *bk, *bv, *bo;      // f32 biases
    int kq;                              // padded K of the Q projection (32 or 128)
};
struct PhaseW {
    MabW mab_in, mab0, mab1;
    const float* I;                      // inducing points [4][128] f32
    const uint16_t *fc_k, *fc_q;         // transposed bf16 [512][128]
    const float *attn_k, *attn_q, *bias; // f32
};

enum { GM_PLAIN = 0, GM_RELURES = 1, GM_TRANS = 2 };

// ---------- WMMA tiled GEMM ----------
// acc = A[rowsPad][kdim] @ Wt^T (+bias);  Wt is [ncols][kdim] (original W transposed).
// GM_PLAIN  : Out[row][col] = acc
// GM_RELURES: Out[row][col] = Res[row][col] + relu(acc)     (Set-Transformer MLP)
// GM_TRANS  : Out[col][row] = acc                           (for Vp^T)
__device__ inline void gemm_tiles(const uint16_t* A, int kdim,
                                  const uint16_t* Wt, const float* bias,
                                  uint16_t* Out, int outStride,
                                  int rowsPad, int ncols,
                                  int mode, const uint16_t* Res, int resStride)
{
    const int wave = threadIdx.x >> 5;
    const int lane = threadIdx.x & 31;
    const int lmod = lane & 15;
    const int lhi  = lane >> 4;
    const int nTiles = ncols >> 4;
    for (int nt = wave; nt < nTiles; nt += 8) {
        const int col = nt * 16 + lmod;
        const uint16_t* Bcol = Wt + (size_t)col * kdim;
        const float bval = bias ? bias[col] : 0.0f;
        for (int mt = 0; mt < rowsPad; mt += 16) {
            const uint16_t* Arow = A + (mt + lmod) * kdim;
            f32x8 acc = {};
            for (int kt = 0; kt < kdim; kt += 32) {
                Frag a, b;
                const u32x4* Ad = (const u32x4*)(Arow + kt);
                const u32x4* Bd = (const u32x4*)(Bcol + kt);
                a.q[0] = Ad[lhi]; a.q[1] = Ad[lhi + 2];
                b.q[0] = Bd[lhi]; b.q[1] = Bd[lhi + 2];
                acc = __builtin_amdgcn_wmma_f32_16x16x32_bf16(
                        false, a.v, false, b.v, (short)0, acc, false, false);
            }
#pragma unroll
            for (int r = 0; r < 8; ++r) {
                const int row = mt + r + 8 * lhi;    // f32 C/D layout
                float v = acc[r] + bval;
                if (mode == GM_RELURES)
                    v = b2f(Res[row * resStride + col]) + fmaxf(v, 0.0f);
                if (mode == GM_TRANS)
                    Out[col * outStride + row] = f2b(v);
                else
                    Out[row * outStride + col] = f2b(v);
            }
        }
    }
}

// ---------- WMMA multi-head attention:  O = Qp + softmax(Qp Kp^T / sqrt(128)) @ Vp ----
// Qp [rq][128], Kp [rk][128], VpT [128][32] (padded cols zeroed),
// sS scratch [4][rq][32] bf16 (scores -> normalized a).
__device__ inline void attention_wmma(const uint16_t* Qp, const uint16_t* Kp,
                                      const uint16_t* VpT, uint16_t* sS,
                                      int nq, int rq, int mk, int rk,
                                      uint16_t* O)
{
    const int wave = threadIdx.x >> 5;
    const int lane = threadIdx.x & 31;
    const int lmod = lane & 15;
    const int lhi  = lane >> 4;
    const int mT = rq >> 4, nT = rk >> 4;
    const float scale = 0.08838834764831843f;  // 1/sqrt(128)

    // scores[h][n][m] = (Qp Kp^T)/sqrt(128): one 16x16x32 WMMA per (head, tile)
    for (int tile = wave; tile < 4 * mT * nT; tile += 8) {
        const int h = tile / (mT * nT);
        const int rem = tile % (mT * nT);
        const int mt = rem / nT, nt = rem % nT;
        Frag a, b;
        const u32x4* Ad = (const u32x4*)(Qp + (mt * 16 + lmod) * 128 + h * 32);
        const u32x4* Bd = (const u32x4*)(Kp + (nt * 16 + lmod) * 128 + h * 32);
        a.q[0] = Ad[lhi]; a.q[1] = Ad[lhi + 2];
        b.q[0] = Bd[lhi]; b.q[1] = Bd[lhi + 2];
        f32x8 acc = {};
        acc = __builtin_amdgcn_wmma_f32_16x16x32_bf16(false, a.v, false, b.v,
                                                      (short)0, acc, false, false);
#pragma unroll
        for (int r = 0; r < 8; ++r) {
            const int row = mt * 16 + r + 8 * lhi;
            sS[(h * rq + row) * 32 + nt * 16 + lmod] = f2b(acc[r] * scale);
        }
    }
    __syncthreads();
    // softmax over m (rows n < nq); zero a[m] for m in [mk,32) so padded K is inert
    for (int i = threadIdx.x; i < nq * 4; i += 256) {
        const int n = i >> 2, h = i & 3;
        uint16_t* row = sS + (h * rq + n) * 32;
        float s[32];
        float mx = -3.0e38f;
        for (int m = 0; m < mk; ++m) { s[m] = b2f(row[m]); mx = fmaxf(mx, s[m]); }
        float sum = 0.f;
        for (int m = 0; m < mk; ++m) { s[m] = __expf(s[m] - mx); sum += s[m]; }
        const float inv = 1.f / sum;
        for (int m = 0; m < mk; ++m) row[m] = f2b(s[m] * inv);
        for (int m = mk; m < 32; ++m) row[m] = 0;
    }
    // zero padded a rows (n in [nq,rq)) so AV leaves O = Qp there (finite)
    for (int i = threadIdx.x; i < 4 * (rq - nq) * 32; i += 256) {
        const int per = (rq - nq) * 32;
        const int h = i / per, rem = i % per;
        sS[(h * rq + nq) * 32 + rem] = 0;
    }
    __syncthreads();
    // O = Qp + a @ Vp (per head, K = 32), residual fused into the D-write
    for (int tile = wave; tile < 4 * mT * 2; tile += 8) {
        const int h = tile / (mT * 2);
        const int rem = tile % (mT * 2);
        const int mt = rem / 2, nt = rem % 2;
        Frag a, b;
        const u32x4* Ad = (const u32x4*)(sS + (h * rq + mt * 16 + lmod) * 32);
        const u32x4* Bd = (const u32x4*)(VpT + (h * 32 + nt * 16 + lmod) * 32);
        a.q[0] = Ad[lhi]; a.q[1] = Ad[lhi + 2];
        b.q[0] = Bd[lhi]; b.q[1] = Bd[lhi + 2];
        f32x8 acc = {};
        acc = __builtin_amdgcn_wmma_f32_16x16x32_bf16(false, a.v, false, b.v,
                                                      (short)0, acc, false, false);
#pragma unroll
        for (int r = 0; r < 8; ++r) {
            const int row = mt * 16 + r + 8 * lhi;
            const int col = h * 32 + nt * 16 + lmod;
            O[row * 128 + col] = f2b(b2f(Qp[row * 128 + col]) + acc[r]);
        }
    }
    __syncthreads();
}

// ---------- one MAB:  Out = O + relu(O@Wo+bo),  O = Qp + MHA(Qp,Kp,Vp) ----------
__device__ inline void mab_run(const MabW& w,
                               const uint16_t* Qin, int nq, int rq,
                               const uint16_t* Kin, int mk, int rk,
                               uint16_t* Qp, uint16_t* Kp, uint16_t* VpT,
                               uint16_t* sS, uint16_t* O, uint16_t* Out)
{
    for (int i = threadIdx.x; i < 128 * 32; i += 256) VpT[i] = 0;  // finite padding
    __syncthreads();
    gemm_tiles(Qin, w.kq, w.Wq, w.bq, Qp, 128, rq, 128, GM_PLAIN, nullptr, 0);
    gemm_tiles(Kin, 128,  w.Wk, w.bk, Kp, 128, rk, 128, GM_PLAIN, nullptr, 0);
    gemm_tiles(Kin, 128,  w.Wv, w.bv, VpT, 32, rk, 128, GM_TRANS, nullptr, 0);
    __syncthreads();
    attention_wmma(Qp, Kp, VpT, sS, nq, rq, mk, rk, O);
    gemm_tiles(O, 128, w.Wo, w.bo, Out, 128, rq, 128, GM_RELURES, O, 128);
    __syncthreads();
}

// ---------- per-set pipeline: gather -> encoder(MAB+ISAB) -> GAT ----------
template<int NSET>
__global__ __launch_bounds__(256) void phase_kernel(
        const float* __restrict__ gfeat, const int* __restrict__ idx,
        const float* __restrict__ wset,  const float* __restrict__ qfeat,
        PhaseW P, float* __restrict__ out)
{
    constexpr int NP = (NSET < 16) ? 16 : NSET;
    constexpr int SZ = NP * 128;
    constexpr int RA = 3 * SZ + 4096 + 4 * NP * 32 + 2048;  // Qp,Kp + VpT + sS + H
    constexpr int RB = NP * 512 + 1552;                     // kft + f32 scratch
    constexpr int REGION = (RA > RB) ? RA : RB;
    __shared__ alignas(16) uint16_t smem[2 * SZ + REGION];

    uint16_t* sK   = smem;                     // gathered K-input, later venc
    uint16_t* sX   = smem + SZ;                // x (mab_in output)
    uint16_t* sQ   = smem + 2 * SZ;            // Q-input, reused as O
    uint16_t* sQp  = sQ  + SZ;
    uint16_t* sKp  = sQp + SZ;
    uint16_t* sVpT = sKp + SZ;                 // [128][32]
    uint16_t* sS   = sVpT + 4096;              // [4][NP][32]
    uint16_t* sH   = sS + 4 * NP * 32;         // [16][128]
    uint16_t* sKft = smem + 2 * SZ;            // [NP][512], overlaps sQ..sH
    float* sQft = (float*)(smem + 2 * SZ + NP * 512);   // in region slack
    float* sEr  = sQft + 512;
    float* sEl  = sEr + 4;
    float* sA   = sEl + NP * 4;

    const int b = blockIdx.x;
    const int t = threadIdx.x;

    // gather member features -> bf16 LDS (+ zero padded rows);
    // set-weights [NSET][16] -> [NP][32] zero-padded
    for (int i = t; i < NSET * 128; i += 256) {
        const int r = i >> 7, c = i & 127;
        sK[i] = f2b(gfeat[(size_t)idx[b * NSET + r] * 128 + c]);
    }
    for (int i = t; i < (NP - NSET) * 128; i += 256) sK[NSET * 128 + i] = 0;
    for (int i = t; i < NP * 32; i += 256) sQ[i] = 0;
    __syncthreads();
    for (int i = t; i < NSET * 16; i += 256) {
        const int r = i >> 4, c = i & 15;
        sQ[r * 32 + c] = f2b(wset[(size_t)b * NSET * 16 + i]);
    }
    __syncthreads();

    // x = MAB(mab_in, W, V)
    mab_run(P.mab_in, sQ, NSET, NP, sK, NSET, NP, sQp, sKp, sVpT, sS, sQ, sX);

    // H = MAB(mab0, I, x)
    for (int i = t; i < 16 * 128; i += 256) sQ[i] = 0;
    __syncthreads();
    for (int i = t; i < 4 * 128; i += 256) sQ[i] = f2b(P.I[i]);
    __syncthreads();
    mab_run(P.mab0, sQ, 4, 16, sX, NSET, NP, sQp, sKp, sVpT, sS, sQ, sH);

    // venc = MAB(mab1, x, H)  -> reuse sK
    mab_run(P.mab1, sX, NSET, NP, sH, 4, 16, sQp, sKp, sVpT, sS, sQ, sK);

    // kft = venc @ fc_k  -> [NP][512] bf16
    gemm_tiles(sK, 128, P.fc_k, nullptr, sKft, 512, NP, 512, GM_PLAIN, nullptr, 0);
    // qft = qfeat[b] @ fc_q  (single row -> scalar VALU)
    for (int c = t; c < 512; c += 256) {
        float acc = 0.f;
        const uint16_t* wc = P.fc_q + (size_t)c * 128;
        const float* qrow = qfeat + (size_t)b * 128;
        for (int k = 0; k < 128; ++k) acc += qrow[k] * b2f(wc[k]);
        sQft[c] = acc;
    }
    __syncthreads();
    if (t < 4) {
        float acc = 0.f;
        for (int d = 0; d < 128; ++d) acc += sQft[t * 128 + d] * P.attn_q[t * 128 + d];
        sEr[t] = acc;
    }
    for (int i = t; i < NSET * 4; i += 256) {
        const int k = i >> 2, h = i & 3;
        float acc = 0.f;
        for (int d = 0; d < 128; ++d)
            acc += b2f(sKft[k * 512 + h * 128 + d]) * P.attn_k[h * 128 + d];
        sEl[i] = acc;
    }
    __syncthreads();
    if (t < 4) {           // softmax over the mailbox axis per head
        const int h = t;
        const float er = sEr[h];
        float vals[NSET];
        float mx = -3.0e38f;
        for (int k = 0; k < NSET; ++k) {
            const float v = fmaxf(sEl[k * 4 + h] + er, 0.f);
            vals[k] = v; mx = fmaxf(mx, v);
        }
        float sum = 0.f;
        for (int k = 0; k < NSET; ++k) { vals[k] = __expf(vals[k] - mx); sum += vals[k]; }
        const float inv = 1.f / sum;
        for (int k = 0; k < NSET; ++k) sA[k * 4 + h] = vals[k] * inv;
    }
    __syncthreads();
    for (int d = t; d < 128; d += 256) {       // (agg + bias).mean over heads
        float acc = 0.f;
        for (int h = 0; h < 4; ++h) {
            float s = P.bias[h * 128 + d];
            for (int k = 0; k < NSET; ++k)
                s += sA[k * 4 + h] * b2f(sKft[k * 512 + h * 128 + d]);
            acc += s;
        }
        out[(size_t)b * 128 + d] = acc * 0.25f;
    }
}

// ---------- weight prep: f32 [K][N] -> transposed bf16 [N][KPAD] (zero-pad K) ----------
__global__ __launch_bounds__(256) void transpose_to_bf16(
        const float* __restrict__ W, uint16_t* __restrict__ out,
        int K, int N, int KPAD)
{
    const int i = blockIdx.x * blockDim.x + threadIdx.x;
    if (i >= N * KPAD) return;
    const int n = i / KPAD, k = i % KPAD;
    out[i] = (k < K) ? f2b(W[(size_t)k * N + n]) : (uint16_t)0;
}

// ---------- host ----------
extern "C" void kernel_launch(void* const* d_in, const int* in_sizes, int n_in,
                              void* d_out, int out_size, void* d_ws, size_t ws_size,
                              hipStream_t stream)
{
    (void)in_sizes; (void)n_in; (void)out_size; (void)ws_size;
    // setup_inputs() insertion order:
    // 0 vfeat, 1 efeat, 2 w_in, 3 w_con, 4 in_src, 5 con_src,
    // 6..13  enc_v.mab_in {Wq,bq,Wk,bk,Wv,bv,Wo,bo}, 14 enc_v.isab.I,
    // 15..22 enc_v.isab.mab0, 23..30 enc_v.isab.mab1,
    // 31..38 enc_e.mab_in, 39 enc_e.isab.I, 40..47 enc_e.isab.mab0, 48..55 enc_e.isab.mab1,
    // 56 fc_ke, 57 fc_qe, 58 attn_ke, 59 attn_qe, 60 bias_e,
    // 61 fc_kv, 62 fc_qv, 63 attn_kv, 64 attn_qv, 65 bias_v
    const float* vfeat   = (const float*)d_in[0];
    const float* efeat   = (const float*)d_in[1];
    const float* w_in    = (const float*)d_in[2];
    const float* w_con   = (const float*)d_in[3];
    const int*   in_src  = (const int*)d_in[4];
    const int*   con_src = (const int*)d_in[5];

    uint16_t* ws = (uint16_t*)d_ws;
    size_t off = 0;
    auto prep = [&](int idx, int K, int N, int KPAD) -> const uint16_t* {
        uint16_t* dst = ws + off;
        const int total = N * KPAD;
        hipLaunchKernelGGL(transpose_to_bf16, dim3((total + 255) / 256), dim3(256), 0,
                           stream, (const float*)d_in[idx], dst, K, N, KPAD);
        off += (size_t)total;
        return dst;
    };
    auto mk_mab = [&](int base, int kq) -> MabW {
        MabW m;
        const int kpad = (kq == 16) ? 32 : 128;
        m.Wq = prep(base + 0, kq, 128, kpad);
        m.bq = (const float*)d_in[base + 1];
        m.Wk = prep(base + 2, 128, 128, 128);
        m.bk = (const float*)d_in[base + 3];
        m.Wv = prep(base + 4, 128, 128, 128);
        m.bv = (const float*)d_in[base + 5];
        m.Wo = prep(base + 6, 128, 128, 128);
        m.bo = (const float*)d_in[base + 7];
        m.kq = kpad;
        return m;
    };

    PhaseW P1, P2;
    P1.mab_in = mk_mab(6, 16);
    P1.I      = (const float*)d_in[14];
    P1.mab0   = mk_mab(15, 128);
    P1.mab1   = mk_mab(23, 128);
    P2.mab_in = mk_mab(31, 16);
    P2.I      = (const float*)d_in[39];
    P2.mab0   = mk_mab(40, 128);
    P2.mab1   = mk_mab(48, 128);
    P1.fc_k   = prep(56, 128, 512, 128);
    P1.fc_q   = prep(57, 128, 512, 128);
    P1.attn_k = (const float*)d_in[58];
    P1.attn_q = (const float*)d_in[59];
    P1.bias   = (const float*)d_in[60];
    P2.fc_k   = prep(61, 128, 512, 128);
    P2.fc_q   = prep(62, 128, 512, 128);
    P2.attn_k = (const float*)d_in[63];
    P2.attn_q = (const float*)d_in[64];
    P2.bias   = (const float*)d_in[65];

    float* out_v = (float*)d_out;                    // [32768][128] vfeat_new
    float* out_e = out_v + (size_t)32768 * 128;      // [8192][128]  efeat_new

    // Phase 1: node -> hyperedge
    hipLaunchKernelGGL((phase_kernel<32>), dim3(8192), dim3(256), 0, stream,
                       vfeat, in_src, w_in, efeat, P1, out_e);
    // Phase 2: hyperedge -> node (reads efeat_new from d_out)
    hipLaunchKernelGGL((phase_kernel<8>), dim3(32768), dim3(256), 0, stream,
                       out_e, con_src, w_con, vfeat, P2, out_v);
}
